// GeodesicAttentionMixer_78795470012855
// MI455X (gfx1250) — matrix operand order repaired and checked by reading, not compiled, not run on hardware
//
#include <hip/hip_runtime.h>

// ---------------------------------------------------------------------------
// CDNA5 / gfx1250 wave32 WMMA types & helpers
// ---------------------------------------------------------------------------
typedef __attribute__((ext_vector_type(16))) __bf16 v16bf;
typedef __attribute__((ext_vector_type(8)))  __bf16 v8bf;
typedef __attribute__((ext_vector_type(8)))  float  v8f;

#define WMMA_BF16(a, b, c) \
  __builtin_amdgcn_wmma_f32_16x16x32_bf16(false, (a), false, (b), (short)0, (c), false, false)

// ---------------------------------------------------------------------------
// prep: convert weights to bf16 (hi/lo split for Wq/Wk, plain bf16 for Wo).
// Storage is row-major [n][k] so stage-1/2 B-fragments are contiguous 32B/lane.
// ---------------------------------------------------------------------------
__global__ void __launch_bounds__(256)
prep_kernel(const float* __restrict__ Wq, const float* __restrict__ Wk,
            const float* __restrict__ Wo,
            __bf16* __restrict__ wqh, __bf16* __restrict__ wql,
            __bf16* __restrict__ wkh, __bf16* __restrict__ wkl,
            __bf16* __restrict__ wo16)
{
  const int idx = blockIdx.x * blockDim.x + threadIdx.x;
  if (idx < 1024 * 1024) wo16[idx] = (__bf16)Wo[idx];
  if (idx < 64 * 64) {
    float a = Wq[idx];
    __bf16 ah = (__bf16)a;
    wqh[idx] = ah;
    wql[idx] = (__bf16)(a - (float)ah);
    float b = Wk[idx];
    __bf16 bh = (__bf16)b;
    wkh[idx] = bh;
    wkl[idx] = (__bf16)(b - (float)bh);
  }
}

// ---------------------------------------------------------------------------
// stage1: one wave per batch.
//   q = x@Wq^T + bq, k = x@Wk^T + bk   (3-term bf16-split WMMA ~ f32 accurate)
//   dist/softmax in f32 VALU via per-wave LDS
//   x_mixed = attn@x, v_mixed = attn@v via bf16 WMMA
//   writes [x_mixed ; v_mixed] as bf16 row-major -> stage-2 A matrix
// ---------------------------------------------------------------------------
#define QP 65  // padded row pitch for q/k LDS tiles

__global__ void __launch_bounds__(128)
stage1_kernel(const float* __restrict__ x, const float* __restrict__ v,
              const float* __restrict__ bq, const float* __restrict__ bk,
              const __bf16* __restrict__ wqh, const __bf16* __restrict__ wql,
              const __bf16* __restrict__ wkh, const __bf16* __restrict__ wkl,
              __bf16* __restrict__ amat, int B)
{
  __shared__ float qs[4][16 * QP];
  __shared__ float ks[4][16 * QP];
  __shared__ float at[4][256];

  const int tid  = threadIdx.x;
  const int wv   = tid >> 5;
  const int lane = tid & 31;
  const int half = lane >> 4;   // 0: lanes 0-15, 1: lanes 16-31
  const int lm   = lane & 15;
  const int b    = blockIdx.x * 4 + wv;

  const float* xb = x + (size_t)b * 1024;
  const float* vb = v + (size_t)b * 1024;

  // ---- A fragments of x (16x32 bf16 layout), hi/lo split, ktile 0 & 1 ----
  // A layout: M = lane%16 ; half t -> K = kt*32 + t + 16*(lane>=16)
  v16bf xh[2], xl[2];
  #pragma unroll
  for (int kt = 0; kt < 2; ++kt) {
    const float* r = xb + lm * 64 + kt * 32 + half * 16;
    alignas(16) float f[16];
    *(float4*)&f[0]  = *(const float4*)(r + 0);
    *(float4*)&f[4]  = *(const float4*)(r + 4);
    *(float4*)&f[8]  = *(const float4*)(r + 8);
    *(float4*)&f[12] = *(const float4*)(r + 12);
    #pragma unroll
    for (int t = 0; t < 16; ++t) {
      __bf16 h = (__bf16)f[t];
      xh[kt][t] = h;
      xl[kt][t] = (__bf16)(f[t] - (float)h);
    }
  }

  // ---- projections: 6 WMMAs per (ntile, matrix) -> q,k tiles into LDS ----
  #pragma unroll
  for (int nt = 0; nt < 4; ++nt) {
    const int col = nt * 16 + lm;
    {
      const __bf16* ph = wqh + col * 64 + half * 16;
      const __bf16* pl = wql + col * 64 + half * 16;
      v16bf h0 = *(const v16bf*)(ph);
      v16bf h1 = *(const v16bf*)(ph + 32);
      v16bf l0 = *(const v16bf*)(pl);
      v16bf l1 = *(const v16bf*)(pl + 32);
      v8f c = {};
      c = WMMA_BF16(xh[0], h0, c); c = WMMA_BF16(xh[1], h1, c);
      c = WMMA_BF16(xh[0], l0, c); c = WMMA_BF16(xh[1], l1, c);
      c = WMMA_BF16(xl[0], h0, c); c = WMMA_BF16(xl[1], h1, c);
      const float bias = bq[col];
      #pragma unroll
      for (int r2 = 0; r2 < 8; ++r2)
        qs[wv][(r2 + half * 8) * QP + col] = c[r2] + bias;
    }
    {
      const __bf16* ph = wkh + col * 64 + half * 16;
      const __bf16* pl = wkl + col * 64 + half * 16;
      v16bf h0 = *(const v16bf*)(ph);
      v16bf h1 = *(const v16bf*)(ph + 32);
      v16bf l0 = *(const v16bf*)(pl);
      v16bf l1 = *(const v16bf*)(pl + 32);
      v8f c = {};
      c = WMMA_BF16(xh[0], h0, c); c = WMMA_BF16(xh[1], h1, c);
      c = WMMA_BF16(xh[0], l0, c); c = WMMA_BF16(xh[1], l1, c);
      c = WMMA_BF16(xl[0], h0, c); c = WMMA_BF16(xl[1], h1, c);
      const float bias = bk[col];
      #pragma unroll
      for (int r2 = 0; r2 < 8; ++r2)
        ks[wv][(r2 + half * 8) * QP + col] = c[r2] + bias;
    }
  }

  // ---- pairwise squared distances + softmax (f32 VALU, lane-pair per row) --
  const int hrow = lane >> 1;
  const int g0   = (lane & 1) * 8;
  float qsq = 0.f;
  #pragma unroll 8
  for (int e = 0; e < 64; ++e) {
    float q = qs[wv][hrow * QP + e];
    qsq += q * q;
  }
  float d[8];
  #pragma unroll
  for (int j = 0; j < 8; ++j) {
    const int g = g0 + j;
    float dot = 0.f, ksq = 0.f;
    #pragma unroll 8
    for (int e = 0; e < 64; ++e) {
      float kk = ks[wv][g * QP + e];
      ksq += kk * kk;
      dot += qs[wv][hrow * QP + e] * kk;
    }
    float dd = qsq + ksq - 2.f * dot;
    d[j] = dd > 0.f ? dd : 0.f;
  }
  float mn = d[0];
  #pragma unroll
  for (int j = 1; j < 8; ++j) mn = fminf(mn, d[j]);
  mn = fminf(mn, __shfl_xor(mn, 1, 32));
  float p[8], s = 0.f;
  #pragma unroll
  for (int j = 0; j < 8; ++j) { p[j] = __expf(mn - d[j]); s += p[j]; }
  s += __shfl_xor(s, 1, 32);
  const float inv = 1.f / s;
  #pragma unroll
  for (int j = 0; j < 8; ++j) at[wv][hrow * 16 + g0 + j] = p[j] * inv;

  // ---- mixing: x_mixed = attn @ x, v_mixed = attn @ v  via bf16 WMMA ------
  // attn is 16x16; pad A to K=32 with zeros (halves with K>=16 are zero).
  v16bf am;
  #pragma unroll
  for (int t = 0; t < 16; ++t) {
    const int K = t + half * 16;
    am[t] = (K < 16) ? (__bf16)at[wv][lm * 16 + K] : (__bf16)0.0f;
  }
  __bf16* outx = amat + (size_t)b * 1024;
  __bf16* outv = amat + ((size_t)B + b) * 1024;
  #pragma unroll
  for (int nt = 0; nt < 4; ++nt) {
    v16bf bx, bv;
    #pragma unroll
    for (int t = 0; t < 16; ++t) {
      const int g = t + half * 16;          // B layout: K = t + 16*half
      float xv = (g < 16) ? xb[g * 64 + nt * 16 + lm] : 0.f;
      float vv = (g < 16) ? vb[g * 64 + nt * 16 + lm] : 0.f;
      bx[t] = (__bf16)xv;
      bv[t] = (__bf16)vv;
    }
    v8f cx = {}; cx = WMMA_BF16(am, bx, cx);
    v8f cv = {}; cv = WMMA_BF16(am, bv, cv);
    #pragma unroll
    for (int r2 = 0; r2 < 8; ++r2) {
      outx[(r2 + half * 8) * 64 + nt * 16 + lm] = (__bf16)cx[r2];
      outv[(r2 + half * 8) * 64 + nt * 16 + lm] = (__bf16)cv[r2];
    }
  }
}

// ---------------------------------------------------------------------------
// stage2: out = A(65536x1024 bf16) @ Wo^T + bo  (f32 out)
// One wave computes a 16x64 C tile; 128 v_wmma_f32_16x16x32_bf16 per wave.
// Wo (2MB bf16) is L2-resident on the 192MB MI455X L2.
// ---------------------------------------------------------------------------
__global__ void __launch_bounds__(256)
stage2_kernel(const __bf16* __restrict__ A, const __bf16* __restrict__ W,
              const float* __restrict__ bo, float* __restrict__ out)
{
  const int tid   = threadIdx.x;
  const int wv    = tid >> 5;
  const int lane  = tid & 31;
  const int half  = lane >> 4;
  const int lm    = lane & 15;
  const int mtile = blockIdx.x * 8 + wv;
  const int nb    = blockIdx.y * 64;

  const __bf16* rowA = A + (size_t)(mtile * 16 + lm) * 1024 + half * 16;
  const __bf16* bp[4];
  #pragma unroll
  for (int nt = 0; nt < 4; ++nt)
    bp[nt] = W + (size_t)(nb + nt * 16 + lm) * 1024 + half * 16;

  v8f acc[4] = {{}, {}, {}, {}};

  #pragma unroll 4
  for (int kt = 0; kt < 32; ++kt) {
    const __bf16* ra = rowA + kt * 32;
    v8bf a0 = *(const v8bf*)(ra);        // halves 0..7  : K = kt*32 + 16*half + t
    v8bf a1 = *(const v8bf*)(ra + 8);    // halves 8..15 : K = ... + 8
    v16bf a;
    #pragma unroll
    for (int t = 0; t < 8; ++t) { a[t] = a0[t]; a[8 + t] = a1[t]; }
    #pragma unroll
    for (int nt = 0; nt < 4; ++nt) {
      v16bf bfrag = *(const v16bf*)(bp[nt] + kt * 32);
      acc[nt] = WMMA_BF16(a, bfrag, acc[nt]);
    }
  }

  float* orow = out + (size_t)mtile * 16 * 1024 + nb + lm;
  #pragma unroll
  for (int nt = 0; nt < 4; ++nt) {
    const float bias = bo[nb + nt * 16 + lm];
    #pragma unroll
    for (int r2 = 0; r2 < 8; ++r2)
      orow[(size_t)(r2 + half * 8) * 1024 + nt * 16] = acc[nt][r2] + bias;
  }
}

// ---------------------------------------------------------------------------
// launch
// ---------------------------------------------------------------------------
extern "C" void kernel_launch(void* const* d_in, const int* in_sizes, int n_in,
                              void* d_out, int out_size, void* d_ws, size_t ws_size,
                              hipStream_t stream)
{
  const float* x  = (const float*)d_in[0];
  const float* v  = (const float*)d_in[1];
  const float* Wq = (const float*)d_in[2];
  const float* bq = (const float*)d_in[3];
  const float* Wk = (const float*)d_in[4];
  const float* bk = (const float*)d_in[5];
  const float* Wo = (const float*)d_in[6];
  const float* bo = (const float*)d_in[7];
  float* out = (float*)d_out;

  const int B = in_sizes[0] / 1024;   // 32768 (x is [B,16,64])

  char* ws = (char*)d_ws;
  __bf16* wqh  = (__bf16*)(ws + 0);
  __bf16* wql  = (__bf16*)(ws + 8192);
  __bf16* wkh  = (__bf16*)(ws + 16384);
  __bf16* wkl  = (__bf16*)(ws + 24576);
  __bf16* wo16 = (__bf16*)(ws + 32768);            // 2 MB
  __bf16* amat = (__bf16*)(ws + (size_t)(4u << 20)); // [2B, 1024] bf16 = 128 MB

  prep_kernel<<<4096, 256, 0, stream>>>(Wq, Wk, Wo, wqh, wql, wkh, wkl, wo16);
  stage1_kernel<<<B / 4, 128, 0, stream>>>(x, v, bq, bk, wqh, wql, wkh, wkl, amat, B);
  dim3 g2((2 * B) / (16 * 8), 16);                 // 512 x 16 blocks, 8 waves each
  stage2_kernel<<<g2, 256, 0, stream>>>(amat, wo16, bo, out);

  (void)n_in; (void)out_size; (void)ws_size;
}
